// KindMed_83597243449581
// MI455X (gfx1250) — matrix operand order repaired and checked by reading, not compiled, not run on hardware
//
#include <hip/hip_runtime.h>
#include <hip/hip_bf16.h>

// ---------------- problem constants (from reference) ----------------
#define DIMV     128
#define N_DIAG   30000
#define N_PRO    20000
#define N_NODES  (N_DIAG + N_PRO)     // 50000
#define N_EDGES  800000
#define N_REL    8
#define TVIS     2
#define KTOT     ((N_REL + 1) * DIMV) // 1152: 8 relations + root
#define LDS_STR  1160                 // padded half stride (16B aligned, bank-skewed)

typedef __attribute__((ext_vector_type(16))) _Float16 v16h;
typedef __attribute__((ext_vector_type(8)))  _Float16 h8;
typedef __attribute__((ext_vector_type(8)))  float    v8f;

// ---------------- utility: zero float4s (grid-sized exactly) ----------------
__global__ void zero_kernel(float4* __restrict__ p, int n4) {
    int i = blockIdx.x * blockDim.x + threadIdx.x;
    if (i < n4) p[i] = make_float4(0.f, 0.f, 0.f, 0.f);
}

// ---------------- gather node features: x = [diag_emb[idx]; pro_emb[idx]] ----------------
__global__ void gather_x_kernel(const int* __restrict__ diag_idx,
                                const int* __restrict__ pro_idx,
                                const float* __restrict__ diag_emb,
                                const float* __restrict__ pro_emb,
                                float* __restrict__ x) {
    int i = blockIdx.x * blockDim.x + threadIdx.x;   // over N_NODES*128
    if (i >= N_NODES * DIMV) return;
    int n = i >> 7, d = i & 127;
    float v;
    if (n < N_DIAG) v = diag_emb[diag_idx[n] * DIMV + d];
    else            v = pro_emb[pro_idx[n - N_DIAG] * DIMV + d];
    x[i] = v;
}

// ---------------- edge scatter: acc[dst,rel,:] += x[src,:], cnt[dst,rel] += 1 ----------------
// one wave32 per edge, 4 floats per lane (float4 read, 4 f32 atomics)
__global__ void rgcn_scatter_kernel(const float* __restrict__ xin,
                                    const int* __restrict__ src,
                                    const int* __restrict__ dst,
                                    const int* __restrict__ et,
                                    float* __restrict__ acc,
                                    float* __restrict__ cnt) {
    int e = blockIdx.x * 8 + (threadIdx.x >> 5);
    if (e >= N_EDGES) return;
    int lane = threadIdx.x & 31;
    int s = src[e], d = dst[e], r = et[e];
    if (lane == 0) atomicAdd(&cnt[d * N_REL + r], 1.0f);
    float4 v = ((const float4*)(xin + (size_t)s * DIMV))[lane];
    float* ap = acc + (size_t)d * (N_REL * DIMV) + r * DIMV + lane * 4;
    atomicAdd(ap + 0, v.x);
    atomicAdd(ap + 1, v.y);
    atomicAdd(ap + 2, v.z);
    atomicAdd(ap + 3, v.w);
}

// ---------------- weight prep: Wcat_t[col][k] (f16), k = r*128+d, r==8 -> root ----------------
__global__ void prep_wcat_kernel(const float* __restrict__ W,    // [8,128,128] (r,d_in,d_out)
                                 const float* __restrict__ root, // [128,128]   (d_in,d_out)
                                 _Float16* __restrict__ wcat) {  // [128][1152]
    int i = blockIdx.x * blockDim.x + threadIdx.x;   // over 128*1152
    if (i >= DIMV * KTOT) return;
    int col = i / KTOT, k = i % KTOT;
    int r = k >> 7, d = k & 127;
    float v = (r < N_REL) ? W[r * DIMV * DIMV + d * DIMV + col] : root[d * DIMV + col];
    wcat[i] = (_Float16)v;
}

// ---------------- RGCN transform: y = relu( [acc/cnt ; x] @ Wcat + bias ) ----------------
// block = 256 threads (8 waves), one 16-node tile; wave w owns output cols [16w,16w+16)
__global__ void rgcn_transform_kernel(const float* __restrict__ xin,
                                      const float* __restrict__ acc,
                                      const float* __restrict__ cnt,
                                      const _Float16* __restrict__ wcat, // [128][1152]
                                      const float* __restrict__ bias,
                                      float* __restrict__ yout) {
    __shared__ _Float16 As[16 * LDS_STR];

    const int n0  = blockIdx.x * 16;
    const int tid = threadIdx.x;

    // stage A tile: 16 rows x 1152 halfs (normalized acc, then x for the root chunk)
    for (int i = tid; i < 16 * KTOT; i += 256) {
        int row = i / KTOT;
        int k   = i % KTOT;
        int n   = n0 + row;
        float v = 0.f;
        if (n < N_NODES) {
            int r = k >> 7, d = k & 127;
            if (r < N_REL) {
                float c = cnt[n * N_REL + r];
                float s = acc[(size_t)n * (N_REL * DIMV) + k];
                v = (c > 0.f) ? (s / c) : 0.f;
            } else {
                v = xin[(size_t)n * DIMV + d];
            }
        }
        As[row * LDS_STR + k] = (_Float16)v;
    }
    __syncthreads();

    const int wave  = tid >> 5;       // 0..7 -> column tile
    const int lane  = tid & 31;
    const int row16 = lane & 15;
    const int hi    = lane >> 4;      // 0: lanes 0-15, 1: lanes 16-31
    const int col   = wave * 16 + row16;

    v8f c = {};
    const _Float16* __restrict__ wcol = wcat + (size_t)col * KTOT;

    for (int k0 = 0; k0 < KTOT; k0 += 32) {
        // A fragment (16x32 f16): lanes 0-15 row=lane K{0-7,16-23}; lanes 16-31 K{8-15,24-31}
        union { v16h v; h8 h[2]; } a;
        const _Float16* arow = &As[row16 * LDS_STR + k0 + hi * 8];
        a.h[0] = *(const h8*)(arow);
        a.h[1] = *(const h8*)(arow + 16);
        // B fragment (32x16 f16): lanes 0-15 col=lane K{0-7,16-23}; lanes 16-31 K{8-15,24-31}
        union { v16h v; h8 h[2]; } b;
        const _Float16* bp = wcol + k0 + hi * 8;
        b.h[0] = *(const h8*)(bp);
        b.h[1] = *(const h8*)(bp + 16);

        c = __builtin_amdgcn_wmma_f32_16x16x32_f16(false, a.v, false, b.v,
                                                   (short)0, c, false, false);
    }

    // C/D layout: VGPR v holds row M = v + 8*hi, col = lane%16
    float bv = bias[col];
    for (int v = 0; v < 8; ++v) {
        int n = n0 + v + hi * 8;
        if (n < N_NODES) {
            float val = c[v] + bv;
            yout[(size_t)n * DIMV + col] = (val > 0.f) ? val : 0.f;
        }
    }
}

// ---------------- readout: cread[d] = sum_n y[n,d] ----------------
__global__ void colsum_kernel(const float* __restrict__ y, float* __restrict__ outv) {
    int d   = threadIdx.x & 127;
    int row = blockIdx.x * 2 + (threadIdx.x >> 7);
    float s = 0.f;
    for (; row < N_NODES; row += gridDim.x * 2)
        s += y[(size_t)row * DIMV + d];
    atomicAdd(&outv[d], s);
}

// ---------------- GRU cell on [1,128] ----------------
__global__ void gru_kernel(const float* __restrict__ x, float* __restrict__ h,
                           const float* __restrict__ Wih, const float* __restrict__ Whh,
                           const float* __restrict__ bih, const float* __restrict__ bhh) {
    __shared__ float xs[DIMV], hs[DIMV];
    int j = threadIdx.x;               // 128 threads
    xs[j] = x[j]; hs[j] = h[j];
    __syncthreads();
    float gir = bih[j], giz = bih[DIMV + j], gin = bih[2 * DIMV + j];
    float ghr = bhh[j], ghz = bhh[DIMV + j], ghn = bhh[2 * DIMV + j];
    for (int k = 0; k < DIMV; ++k) {
        float xv = xs[k], hv = hs[k];
        gir += Wih[j * DIMV + k] * xv;
        giz += Wih[(DIMV + j) * DIMV + k] * xv;
        gin += Wih[(2 * DIMV + j) * DIMV + k] * xv;
        ghr += Whh[j * DIMV + k] * hv;
        ghz += Whh[(DIMV + j) * DIMV + k] * hv;
        ghn += Whh[(2 * DIMV + j) * DIMV + k] * hv;
    }
    float r  = 1.f / (1.f + __expf(-(gir + ghr)));
    float z  = 1.f / (1.f + __expf(-(giz + ghz)));
    float nn = tanhf(gin + r * ghn);
    float hn = (1.f - z) * nn + z * hs[j];
    __syncthreads();
    h[j] = hn;
}

// ---------------- head: out = relu(h@Wo1^T+bo1)@Wo2^T+bo2 ----------------
__global__ void head_kernel(const float* __restrict__ h,
                            const float* __restrict__ Wo1, const float* __restrict__ bo1,
                            const float* __restrict__ Wo2, const float* __restrict__ bo2,
                            float* __restrict__ out) {
    __shared__ float hsm[DIMV];
    __shared__ float hid[2 * DIMV];
    int j = threadIdx.x;               // 256 threads
    if (j < DIMV) hsm[j] = h[j];
    __syncthreads();
    float s = bo1[j];
    for (int k = 0; k < DIMV; ++k) s += Wo1[j * DIMV + k] * hsm[k];
    hid[j] = (s > 0.f) ? s : 0.f;
    __syncthreads();
    if (j < 150) {
        float o = bo2[j];
        for (int k = 0; k < 2 * DIMV; ++k) o += Wo2[j * 2 * DIMV + k] * hid[k];
        out[j] = o;
    }
}

// ---------------- host orchestration ----------------
extern "C" void kernel_launch(void* const* d_in, const int* in_sizes, int n_in,
                              void* d_out, int out_size, void* d_ws, size_t ws_size,
                              hipStream_t stream) {
    (void)in_sizes; (void)n_in; (void)out_size; (void)ws_size;

    const int*   c_diag_idx = (const int*)  d_in[0];   // [2,30000]
    const int*   c_pro_idx  = (const int*)  d_in[1];   // [2,20000]
    const int*   c_edge_idx = (const int*)  d_in[2];   // [2,2,800000]
    const int*   c_edge_typ = (const int*)  d_in[3];   // [2,800000]
    const float* diag_emb   = (const float*)d_in[4];
    const float* pro_emb    = (const float*)d_in[5];
    const float* W1    = (const float*)d_in[6];
    const float* root1 = (const float*)d_in[7];
    const float* b1    = (const float*)d_in[8];
    const float* W2    = (const float*)d_in[9];
    const float* root2 = (const float*)d_in[10];
    const float* b2    = (const float*)d_in[11];
    const float* gWih  = (const float*)d_in[12];
    const float* gWhh  = (const float*)d_in[13];
    const float* gbih  = (const float*)d_in[14];
    const float* gbhh  = (const float*)d_in[15];
    const float* Wo1   = (const float*)d_in[16];
    const float* bo1   = (const float*)d_in[17];
    const float* Wo2   = (const float*)d_in[18];
    const float* bo2   = (const float*)d_in[19];
    float* out = (float*)d_out;

    // ---- workspace carve (bytes, 256-aligned) ----
    char* ws = (char*)d_ws;
    const size_t accB   = (size_t)N_NODES * N_REL * DIMV * 4;  // 204.8 MB
    const size_t cntB   = (size_t)N_NODES * N_REL * 4;         // 1.6 MB (directly after acc)
    const size_t xB     = (size_t)N_NODES * DIMV * 4;          // 25.6 MB
    float*     acc   = (float*)ws;                 // acc and cnt contiguous -> single zero pass
    float*     cnt   = (float*)(ws + accB);
    float*     xbuf  = (float*)(ws + accB + cntB);
    float*     ybuf  = (float*)(ws + accB + cntB + xB);
    _Float16*  wcat1 = (_Float16*)(ws + accB + cntB + 2 * xB);
    _Float16*  wcat2 = wcat1 + (size_t)DIMV * KTOT;
    float*     cread = (float*)((char*)(wcat2 + (size_t)DIMV * KTOT) + 256);
    float*     hbuf  = cread + 128;

    const int accCntF4 = (int)((accB + cntB) / 16);            // float4 count

    // ---- one-time weight conversion (f32 -> f16, transposed concat layout) ----
    {
        int n = DIMV * KTOT, blk = (n + 255) / 256;
        prep_wcat_kernel<<<blk, 256, 0, stream>>>(W1, root1, wcat1);
        prep_wcat_kernel<<<blk, 256, 0, stream>>>(W2, root2, wcat2);
    }
    // zero h only (128 floats; cread is zeroed each step)
    zero_kernel<<<1, 64, 0, stream>>>((float4*)hbuf, 32);

    for (int t = 0; t < TVIS; ++t) {
        const int* src = c_edge_idx + (size_t)t * 2 * N_EDGES;
        const int* dst = src + N_EDGES;
        const int* et  = c_edge_typ + (size_t)t * N_EDGES;

        // gather node features
        {
            int n = N_NODES * DIMV, blk = (n + 255) / 256;
            gather_x_kernel<<<blk, 256, 0, stream>>>(c_diag_idx + t * N_DIAG,
                                                     c_pro_idx  + t * N_PRO,
                                                     diag_emb, pro_emb, xbuf);
        }

        // ---- layer 1: x -> y ----
        zero_kernel<<<(accCntF4 + 255) / 256, 256, 0, stream>>>((float4*)acc, accCntF4);
        rgcn_scatter_kernel<<<(N_EDGES + 7) / 8, 256, 0, stream>>>(xbuf, src, dst, et, acc, cnt);
        rgcn_transform_kernel<<<(N_NODES + 15) / 16, 256, 0, stream>>>(xbuf, acc, cnt, wcat1, b1, ybuf);

        // ---- layer 2: y -> x ----
        zero_kernel<<<(accCntF4 + 255) / 256, 256, 0, stream>>>((float4*)acc, accCntF4);
        rgcn_scatter_kernel<<<(N_EDGES + 7) / 8, 256, 0, stream>>>(ybuf, src, dst, et, acc, cnt);
        rgcn_transform_kernel<<<(N_NODES + 15) / 16, 256, 0, stream>>>(ybuf, acc, cnt, wcat2, b2, xbuf);

        // ---- readout + GRU ----
        zero_kernel<<<1, 32, 0, stream>>>((float4*)cread, 32);
        colsum_kernel<<<512, 256, 0, stream>>>(xbuf, cread);
        gru_kernel<<<1, 128, 0, stream>>>(cread, hbuf, gWih, gWhh, gbih, gbhh);
    }

    head_kernel<<<1, 256, 0, stream>>>(hbuf, Wo1, bo1, Wo2, bo2, out);
}
